// WindowAttention1D_88218628260826
// MI455X (gfx1250) — matrix operand order repaired and checked by reading, not compiled, not run on hardware
//
#include <hip/hip_runtime.h>

#define DIM   512
#define WIN   64
#define HEADS 16
#define HD    32
#define LDX   520   // padded row stride (halfs) for x/q/k tiles: 260 dwords -> conflict-free
#define LDV   68    // padded row stride (halfs) for transposed V

typedef _Float16 half4_t  __attribute__((ext_vector_type(4)));
typedef _Float16 half8_t  __attribute__((ext_vector_type(8)));
typedef _Float16 v16h     __attribute__((ext_vector_type(16)));
typedef float    v8f      __attribute__((ext_vector_type(8)));
typedef float    float4_t __attribute__((ext_vector_type(4)));

static __device__ __forceinline__ v8f wmma_f16(v16h a, v16h b, v8f c) {
  // D = A(16x32 f16) * B(32x16 f16) + C(16x16 f32)
  return __builtin_amdgcn_wmma_f32_16x16x32_f16(false, a, false, b, (short)0, c,
                                                false, false);
}
static __device__ __forceinline__ v16h cat88(half8_t lo, half8_t hi) {
  return __builtin_shufflevector(lo, hi, 0,1,2,3,4,5,6,7,8,9,10,11,12,13,14,15);
}
static __device__ __forceinline__ half8_t cat44(half4_t lo, half4_t hi) {
  return __builtin_shufflevector(lo, hi, 0,1,2,3,4,5,6,7);
}

// ---------------- fp32 -> fp16 weight conversion (once per launch) ----------
__global__ void cvt_f32_to_f16(const float* __restrict__ src,
                               _Float16* __restrict__ dst, int n4) {
  int i = blockIdx.x * blockDim.x + threadIdx.x;
  int stride = gridDim.x * blockDim.x;
  const float4_t* s4 = (const float4_t*)src;
  half4_t* d4 = (half4_t*)dst;
  for (; i < n4; i += stride) {
    float4_t v = s4[i];
    half4_t h = {(_Float16)v.x, (_Float16)v.y, (_Float16)v.z, (_Float16)v.w};
    d4[i] = h;
  }
}

// ---------------- fused window attention: one block per window --------------
__global__ __launch_bounds__(256) void win_attn(
    const float* __restrict__ x, const float* __restrict__ qkv_b,
    const float* __restrict__ proj_b, const float* __restrict__ bias_tb,
    const _Float16* __restrict__ wqkv, const _Float16* __restrict__ wproj,
    float* __restrict__ out) {
  // Single LDS arena; regions reused across phases.
  // [sX: 64*LDX f16][sQ: 64*LDX f16][sK: 64*LDX f16][sVT: 512*LDV f16]
  // After QKV phase: sX region reused as per-wave P buffers (+bias column),
  //                  sK region reused as attention output O.
  __shared__ __align__(16) char smem[(3 * WIN * LDX + DIM * LDV) * 2];
  _Float16* sX  = (_Float16*)smem;
  _Float16* sQ  = sX + WIN * LDX;
  _Float16* sK  = sQ + WIN * LDX;
  _Float16* sVT = sK + WIN * LDX;

  const int tid = threadIdx.x;
  const int wv  = tid >> 5;   // wave id 0..7
  const int ln  = tid & 31;   // lane
  const int lh  = ln & 15;    // lane within half-wave
  const int hi  = ln >> 4;    // half-wave select
  const int b   = blockIdx.x;

  // ---- phase 0: load x tile [64,512] f32 -> f16 LDS (padded rows) ----------
  {
    const float4_t* xg = (const float4_t*)(x + (size_t)b * (WIN * DIM));
#pragma unroll
    for (int i = 0; i < 32; ++i) {
      int idx = tid + i * 256;          // float4 index: 128 per row
      int row = idx >> 7, c4 = idx & 127;
      float4_t v = xg[idx];
      half4_t h = {(_Float16)v.x, (_Float16)v.y, (_Float16)v.z, (_Float16)v.w};
      *(half4_t*)(sX + row * LDX + c4 * 4) = h;
    }
  }
  __syncthreads();

  // ---- phase 1: QKV GEMM  C[64,1536] = X * Wqkv^T + b ----------------------
  // wave w owns n-tiles w, w+8, ..., w+88  (96 tiles of 16 cols)
  for (int j = 0; j < 12; ++j) {
    const int ntile = wv + 8 * j;
    const int ncol  = ntile * 16 + lh;                  // lane's output column
    const _Float16* brow = wqkv + (size_t)ncol * DIM + hi * 16;
    v8f acc[4] = {};
    for (int kt = 0; kt < 16; ++kt) {
      const _Float16* bp = brow + kt * 32;
      v16h B = cat88(*(const half8_t*)bp, *(const half8_t*)(bp + 8));
#pragma unroll
      for (int mt = 0; mt < 4; ++mt) {
        const _Float16* ap = sX + (mt * 16 + lh) * LDX + kt * 32 + hi * 16;
        v16h A = cat88(*(const half8_t*)ap, *(const half8_t*)(ap + 8));
        acc[mt] = wmma_f16(A, B, acc[mt]);
      }
    }
    const float bias = qkv_b[ncol];
    if (ntile < 32) {                                   // q (fold 1/sqrt(32))
      const float sc = 0.17677669529663687f;
#pragma unroll
      for (int mt = 0; mt < 4; ++mt)
#pragma unroll
        for (int r = 0; r < 8; ++r)
          sQ[(mt * 16 + r + hi * 8) * LDX + ncol] =
              (_Float16)((acc[mt][r] + bias) * sc);
    } else if (ntile < 64) {                            // k
      const int c = ncol - 512;
#pragma unroll
      for (int mt = 0; mt < 4; ++mt)
#pragma unroll
        for (int r = 0; r < 8; ++r)
          sK[(mt * 16 + r + hi * 8) * LDX + c] = (_Float16)(acc[mt][r] + bias);
    } else {                                            // v, stored transposed
      const int c = ncol - 1024;
#pragma unroll
      for (int mt = 0; mt < 4; ++mt)
#pragma unroll
        for (int r = 0; r < 8; ++r)
          sVT[c * LDV + (mt * 16 + r + hi * 8)] = (_Float16)(acc[mt][r] + bias);
    }
  }
  __syncthreads();

  // ---- phase 2: per-head attention (wave w -> heads 2w, 2w+1) --------------
  _Float16* pbuf = sX + wv * (WIN * WIN);   // per-wave 8KB P buffer in dead x
  float* bcol = (float*)pbuf;               // bias-table column overlay
#pragma unroll 1
  for (int hh = 0; hh < 2; ++hh) {
    const int h = wv * 2 + hh;
    for (int d = ln; d < 2 * WIN - 1; d += 32) bcol[d] = bias_tb[d * HEADS + h];

    // S = q_h * k_h^T  (K = HD = 32 -> one WMMA per 16x16 tile)
    v16h Aq[4], Bk[4];
#pragma unroll
    for (int mt = 0; mt < 4; ++mt) {
      const _Float16* ap = sQ + (mt * 16 + lh) * LDX + h * HD + hi * 16;
      Aq[mt] = cat88(*(const half8_t*)ap, *(const half8_t*)(ap + 8));
    }
#pragma unroll
    for (int nt = 0; nt < 4; ++nt) {
      const _Float16* bp = sK + (nt * 16 + lh) * LDX + h * HD + hi * 16;
      Bk[nt] = cat88(*(const half8_t*)bp, *(const half8_t*)(bp + 8));
    }
    v8f S[4][4] = {};
#pragma unroll
    for (int mt = 0; mt < 4; ++mt)
#pragma unroll
      for (int nt = 0; nt < 4; ++nt) S[mt][nt] = wmma_f16(Aq[mt], Bk[nt], S[mt][nt]);

    // relative-position bias: rel = (i - j) + 63, gathered from LDS column
#pragma unroll
    for (int mt = 0; mt < 4; ++mt)
#pragma unroll
      for (int nt = 0; nt < 4; ++nt) {
        const int base = 16 * (mt - nt) + 63 + hi * 8 - lh;
#pragma unroll
        for (int r = 0; r < 8; ++r) S[mt][nt][r] += bcol[base + r];
      }

    // softmax over 64 cols: 4 in-lane tiles + shfl_xor across the 16-lane half
#pragma unroll
    for (int mt = 0; mt < 4; ++mt)
#pragma unroll
      for (int r = 0; r < 8; ++r) {
        float mx = fmaxf(fmaxf(S[mt][0][r], S[mt][1][r]),
                         fmaxf(S[mt][2][r], S[mt][3][r]));
        mx = fmaxf(mx, __shfl_xor(mx, 1, 32));
        mx = fmaxf(mx, __shfl_xor(mx, 2, 32));
        mx = fmaxf(mx, __shfl_xor(mx, 4, 32));
        mx = fmaxf(mx, __shfl_xor(mx, 8, 32));
        float e0 = __expf(S[mt][0][r] - mx), e1 = __expf(S[mt][1][r] - mx);
        float e2 = __expf(S[mt][2][r] - mx), e3 = __expf(S[mt][3][r] - mx);
        float sum = (e0 + e1) + (e2 + e3);
        sum += __shfl_xor(sum, 1, 32);
        sum += __shfl_xor(sum, 2, 32);
        sum += __shfl_xor(sum, 4, 32);
        sum += __shfl_xor(sum, 8, 32);
        const float inv = 1.0f / sum;
        S[mt][0][r] = e0 * inv; S[mt][1][r] = e1 * inv;
        S[mt][2][r] = e2 * inv; S[mt][3][r] = e3 * inv;
      }

    // C-layout -> LDS (f16), then reload as WMMA A operand
#pragma unroll
    for (int mt = 0; mt < 4; ++mt)
#pragma unroll
      for (int nt = 0; nt < 4; ++nt)
#pragma unroll
        for (int r = 0; r < 8; ++r)
          pbuf[(mt * 16 + r + hi * 8) * WIN + nt * 16 + lh] =
              (_Float16)S[mt][nt][r];

    // O_h = P * V_h ; write f16 into dead k region (row-major [64,512])
#pragma unroll
    for (int mt = 0; mt < 4; ++mt)
#pragma unroll
      for (int nt2 = 0; nt2 < 2; ++nt2) {
        v8f o = {};
#pragma unroll
        for (int kt = 0; kt < 2; ++kt) {
          const _Float16* ap = pbuf + (mt * 16 + lh) * WIN + kt * 32 + hi * 16;
          v16h A = cat88(*(const half8_t*)ap, *(const half8_t*)(ap + 8));
          const _Float16* bp =
              sVT + (h * HD + nt2 * 16 + lh) * LDV + kt * 32 + hi * 16;
          half8_t blo = cat44(*(const half4_t*)bp, *(const half4_t*)(bp + 4));
          half8_t bhi = cat44(*(const half4_t*)(bp + 8), *(const half4_t*)(bp + 12));
          o = wmma_f16(A, cat88(blo, bhi), o);
        }
#pragma unroll
        for (int r = 0; r < 8; ++r)
          sK[(mt * 16 + r + hi * 8) * LDX + h * HD + nt2 * 16 + lh] =
              (_Float16)o[r];
      }
  }
  __syncthreads();

  // ---- phase 3: projection  Y[64,512] = O * Wproj^T + b -> global f32 ------
  float* orow = out + (size_t)b * (WIN * DIM);
  for (int j = 0; j < 4; ++j) {
    const int ntile = wv + 8 * j;
    const int ncol  = ntile * 16 + lh;
    const _Float16* brow = wproj + (size_t)ncol * DIM + hi * 16;
    v8f acc[4] = {};
    for (int kt = 0; kt < 16; ++kt) {
      const _Float16* bp = brow + kt * 32;
      v16h B = cat88(*(const half8_t*)bp, *(const half8_t*)(bp + 8));
#pragma unroll
      for (int mt = 0; mt < 4; ++mt) {
        const _Float16* ap = sK + (mt * 16 + lh) * LDX + kt * 32 + hi * 16;
        v16h A = cat88(*(const half8_t*)ap, *(const half8_t*)(ap + 8));
        acc[mt] = wmma_f16(A, B, acc[mt]);
      }
    }
    const float bias = proj_b[ncol];
#pragma unroll
    for (int mt = 0; mt < 4; ++mt)
#pragma unroll
      for (int r = 0; r < 8; ++r)
        orow[(mt * 16 + r + hi * 8) * DIM + ncol] = acc[mt][r] + bias;
  }
}

extern "C" void kernel_launch(void* const* d_in, const int* in_sizes, int n_in,
                              void* d_out, int out_size, void* d_ws,
                              size_t ws_size, hipStream_t stream) {
  (void)in_sizes; (void)n_in; (void)out_size; (void)ws_size;
  const float* x       = (const float*)d_in[0];
  const float* qkv_w   = (const float*)d_in[1];
  const float* qkv_b   = (const float*)d_in[2];
  const float* proj_w  = (const float*)d_in[3];
  const float* proj_b  = (const float*)d_in[4];
  const float* bias_tb = (const float*)d_in[5];
  // d_in[6] (rel_index) unused: rel = (i - j) + 63 computed analytically.

  _Float16* wqkv  = (_Float16*)d_ws;            // [1536,512] f16 (1.5 MB)
  _Float16* wproj = wqkv + 1536 * 512;          // [512,512]  f16 (0.5 MB)

  cvt_f32_to_f16<<<512, 256, 0, stream>>>(qkv_w, wqkv, (1536 * 512) / 4);
  cvt_f32_to_f16<<<256, 256, 0, stream>>>(proj_w, wproj, (512 * 512) / 4);
  win_attn<<<4096, 256, 0, stream>>>(x, qkv_b, proj_b, bias_tb, wqkv, wproj,
                                     (float*)d_out);
}
